// DualMotionGAN_62191126446592
// MI455X (gfx1250) — compile-verified, matrix operand order checked by reading, not run on hardware
//
#include <hip/hip_runtime.h>

typedef _Float16 f16;
typedef __attribute__((ext_vector_type(8)))  _Float16 v8h;
typedef __attribute__((ext_vector_type(16))) _Float16 v16h;
typedef __attribute__((ext_vector_type(8)))  float    v8f;

enum { ACT_NONE=0, ACT_LEAKY=1, ACT_RELU=2, ACT_SIG=3, ACT_TANH=4 };

// ---------------------------------------------------------------------------
// Implicit-GEMM conv via V_WMMA_F32_16X16X32_F16 with double-buffered LDS
// staging of the packed-B tile.
//   A (M x K): M = N*OH*OW output pixels, K = KH*KW*IC, k ordered (ky,kx,ic).
//   Input is pre-padded NHWC f16; IC % 32 == 0 for every WMMA layer, so each
//   32-wide k-chunk is two contiguous 16B halves per lane (exact ISA 16-bit
//   A-fragment layout: lanes<16 hold K{0-7,16-23}, lanes>=16 hold K{8-15,24-31}).
//   B is pre-packed in fragment order: ((kt*NT + nt)*32 + lane)*16 + j halves;
//   for a fixed kt the NTW tiles are contiguous -> one cooperative 256-thread
//   copy (uint4 each) stages the whole k-step tile into LDS once per block
//   instead of 8 redundant per-wave global reads.
//   Wave tile: MTW=2 16-row M subtiles x NTW 16-col N tiles (32x64/wave,
//   8 WMMAs per K-step, each B fragment reused across both M subtiles).
//   Block = 256 threads = 8 waves => 256-row macro tile; every conv M here is
//   a multiple of 256 and every OW a multiple of 16 (no tails; all 16 rows of
//   a subtile share (n,oy)).
// ---------------------------------------------------------------------------
template<int ACT, typename OT, int MTW, int NTW>
__global__ __launch_bounds__(256) void conv_wmma(
    const f16* __restrict__ in, const f16* __restrict__ wpk,
    const float* __restrict__ bias, OT* __restrict__ out,
    int IHp, int IWp, int IC,
    int OH, int OW, int OC, int OCpad,
    int KH, int KW, int stride,
    int OHp, int OWp, int padO)
{
  __shared__ f16 lb[2][NTW * 512];               // NTW KB per stage buffer

  const int tid  = threadIdx.x;
  const int lane = tid & 31;
  const int wave = tid >> 5;
  const int rowBase0 = blockIdx.x * (128 * MTW) + wave * (16 * MTW);
  const int kk0 = (lane & 16) ? 8 : 0;           // ISA A/B-fragment K split
  const int NT = OCpad >> 4;
  const int nt0 = blockIdx.y * NTW;
  const int l15 = lane & 15;

  long inBase[MTW];
  #pragma unroll
  for (int s = 0; s < MTW; ++s) {
    int rb = rowBase0 + 16 * s;
    int t2 = rb / OW; int oxB = rb - t2 * OW;    // 16 rows share (n,oy)
    int n  = t2 / OH; int oy  = t2 - n * OH;
    inBase[s] = ((long)(n * IHp + oy * stride) * IWp + (oxB + l15) * stride) * IC + kk0;
  }

  v8f acc[MTW][NTW];
  #pragma unroll
  for (int s = 0; s < MTW; ++s)
    #pragma unroll
    for (int j = 0; j < NTW; ++j)
      #pragma unroll
      for (int e = 0; e < 8; ++e) acc[s][j][e] = 0.f;

  const int KT = KH * KW * (IC >> 5);
  const f16* bsrc = wpk + (long)nt0 * 512;       // k-step tile group (NTW KB)

  // stage k-step 0
  if (NTW == 4) ((uint4*)lb[0])[tid] = ((const uint4*)bsrc)[tid];
  else ((unsigned int*)lb[0])[tid] = ((const unsigned int*)bsrc)[tid];
  __syncthreads();

  int kt = 0;
  for (int ky = 0; ky < KH; ++ky)
  for (int kx = 0; kx < KW; ++kx) {
    const long roff = ((long)ky * IWp + kx) * IC;
    // keep L0 warm for the next im2col tap (lowers to global_prefetch_b8)
    __builtin_prefetch(in + inBase[0] + roff + IC, 0, 0);
    for (int icc = 0; icc < IC; icc += 32, ++kt) {
      const int cur = kt & 1;
      // stage next k-step tile into the other buffer (uniform predicate)
      if (kt + 1 < KT) {
        const f16* nsrc = bsrc + (long)NT * 512;
        if (NTW == 4) ((uint4*)lb[1 - cur])[tid] = ((const uint4*)nsrc)[tid];
        else ((unsigned int*)lb[1 - cur])[tid] = ((const unsigned int*)nsrc)[tid];
      }
      bsrc += (long)NT * 512;

      v16h a[MTW];
      #pragma unroll
      for (int s = 0; s < MTW; ++s) {
        const f16* ap = in + inBase[s] + roff + icc;
        v8h lo = *(const v8h*)ap;                // K = kk0 + 0..7
        v8h hi = *(const v8h*)(ap + 16);         // K = kk0 + 16..23
        #pragma unroll
        for (int i = 0; i < 8; ++i) { a[s][i] = lo[i]; a[s][i + 8] = hi[i]; }
      }
      #pragma unroll
      for (int j = 0; j < NTW; ++j) {
        v16h b = *(const v16h*)&lb[cur][(j * 32 + lane) * 16];
        #pragma unroll
        for (int s = 0; s < MTW; ++s)
          acc[s][j] = __builtin_amdgcn_wmma_f32_16x16x32_f16(
              false, a[s], false, b, (short)0, acc[s][j], false, false);
      }
      __syncthreads();   // readers done with lb[cur]; lb[1-cur] now visible
    }
  }

  // D layout: col = lane&15; VGPR e holds row e (lanes<16) / e+8 (lanes>=16)
  const int mHi = (lane & 16) >> 1;
  const int col = lane & 15;
  #pragma unroll
  for (int s = 0; s < MTW; ++s) {
    int rb = rowBase0 + 16 * s;
    int t2 = rb / OW; int oxB = rb - t2 * OW;
    int n  = t2 / OH; int oy  = t2 - n * OH;
    long ob = ((long)(n * OHp + oy + padO) * OWp + (oxB + mHi + padO)) * OC;
    #pragma unroll
    for (int j = 0; j < NTW; ++j) {
      int oc = (nt0 + j) * 16 + col;
      if (oc >= OC) continue;                    // OCpad padding lanes
      float bv = bias ? bias[oc] : 0.f;
      #pragma unroll
      for (int e = 0; e < 8; ++e) {
        float v = acc[s][j][e] + bv;
        if      (ACT == ACT_LEAKY) v = v > 0.f ? v : 0.2f * v;
        else if (ACT == ACT_RELU)  v = v > 0.f ? v : 0.f;
        else if (ACT == ACT_SIG)   v = 1.f / (1.f + expf(-v));
        else if (ACT == ACT_TANH)  v = tanhf(v);
        out[ob + (long)e * OC + oc] = (OT)v;
      }
    }
  }
}

// Pack OIHW f32 weights into WMMA B-fragment order (f16), k = (ky,kx,ic).
// dst[((kt*NT + nt)*32 + lane)*16 + j] = w[n=nt*16+(lane&15)][k=kt*32+kk],
// kk = (j<8?0:16) + (lane>=16?8:0) + (j&7).  OC zero-padded to OCpad.
__global__ void pack_wmma_b(const float* __restrict__ w, f16* __restrict__ dst,
                            int OC, int IC, int KH, int KW, int OCpad)
{
  long idx = (long)blockIdx.x * 256 + threadIdx.x;
  int K = KH * KW * IC;
  long total = (long)(K >> 5) * (OCpad >> 4) * 512;
  if (idx >= total) return;
  int j = idx & 15;
  int lane = (idx >> 4) & 31;
  long tile = idx >> 9;
  int NT = OCpad >> 4;
  int nt = (int)(tile % NT); int kt = (int)(tile / NT);
  int kk = ((j >> 3) << 4) + ((lane >> 4) << 3) + (j & 7);
  int k = kt * 32 + kk;
  int nn = nt * 16 + (lane & 15);
  f16 v = (f16)0.f;
  if (nn < OC) {
    int ic = k % IC; int q = k / IC; int kx = q % KW; int ky = q / KW;
    v = (f16)w[(((long)nn * IC + ic) * KH + ky) * KW + kx];
  }
  dst[idx] = v;
}

// Encoder layer 1 (IC=3, too thin for WMMA; memory-bound).
// x (B=4,C=3,T=4,256,256) f32 -> e1p (16,130,130,64) f16, leaky 0.2, pad interior.
__global__ void conv1_enc(const float* __restrict__ x, const float* __restrict__ w,
                          f16* __restrict__ out)
{
  long idx = (long)blockIdx.x * 256 + threadIdx.x;   // 16*128*128*64
  int oc = idx & 63;
  int ox = (idx >> 6) & 127;
  int oy = (idx >> 13) & 127;
  int n  = (int)(idx >> 20);
  int b = n >> 2, tt = n & 3;
  float s = 0.f;
  for (int ky = 0; ky < 4; ++ky) {
    int iy = oy * 2 - 1 + ky; if ((unsigned)iy >= 256u) continue;
    for (int kx = 0; kx < 4; ++kx) {
      int ix = ox * 2 - 1 + kx; if ((unsigned)ix >= 256u) continue;
      for (int ic = 0; ic < 3; ++ic)
        s += x[((((long)(b * 3 + ic) * 4 + tt) * 256 + iy) * 256) + ix]
           * w[((oc * 3 + ic) * 4 + ky) * 4 + kx];
    }
  }
  s = s > 0.f ? s : 0.2f * s;
  out[(((long)(n * 130) + oy + 1) * 130 + ox + 1) * 64 + oc] = (f16)s;
}

// Build padded LSTM conv input: xh (4,18,18,1024) interior = [xt ; h_prev]
__global__ void build_xh(const f16* __restrict__ xt, long bStride,
                         const f16* __restrict__ h, f16* __restrict__ xh)
{
  int idx = blockIdx.x * 256 + threadIdx.x;          // 4*16*16*1024
  int cc = idx & 1023;
  int x = (idx >> 10) & 15;
  int y = (idx >> 14) & 15;
  int b = idx >> 18;
  f16 v = (cc < 512) ? xt[(long)b * bStride + ((long)(y * 16) + x) * 512 + cc]
                     : h[(((long)(b * 16) + y) * 16 + x) * 512 + (cc - 512)];
  xh[(((long)(b * 18) + y + 1) * 18 + (x + 1)) * 1024 + cc] = v;
}

// LSTM pointwise: gates (4,16,16,2048) f32 [i|f|g|o] -> c (f32), h (f16 x2)
__global__ void lstm_point(const float* __restrict__ g, float* __restrict__ c,
                           f16* __restrict__ hbuf, f16* __restrict__ seqOut)
{
  int idx = blockIdx.x * 256 + threadIdx.x;          // 4*16*16*512
  int ch = idx & 511;
  long pix = idx >> 9;
  const float* gp = g + pix * 2048;
  float gi = gp[ch], gf = gp[512 + ch], gg = gp[1024 + ch], go = gp[1536 + ch];
  float si = 1.f / (1.f + expf(-gi));
  float sf = 1.f / (1.f + expf(-gf));
  float so = 1.f / (1.f + expf(-go));
  float cn = sf * c[idx] + si * tanhf(gg);
  c[idx] = cn;
  float h = so * tanhf(cn);
  hbuf[idx] = (f16)h;
  seqOut[idx] = (f16)h;
}

// Deconv pre-step: scatter g (N,H,W,C) into zeroed dilated+padded buffer
// db (N, 2H+2, 2H+2, C) at (1+2y, 1+2x)  [lhs_dilation=2, pad (1,2)]
__global__ void scatter_dil(const f16* __restrict__ g, f16* __restrict__ db,
                            int H, int C)
{
  long idx = (long)blockIdx.x * 256 + threadIdx.x;   // 4*H*H*C
  long tot = (long)4 * H * H * C;
  if (idx >= tot) return;
  int c = (int)(idx % C); long p = idx / C;
  int x = (int)(p % H); p /= H;
  int y = (int)(p % H); int n = (int)(p / H);
  int S = 2 * H + 2;
  db[(((long)n * S + 1 + 2 * y) * S + (1 + 2 * x)) * C + c] = g[idx];
}

// NHWC f32 -> NCHW f32 writer
__global__ void write_nchw(const float* __restrict__ src, float* __restrict__ dst, int C)
{
  long idx = (long)blockIdx.x * 256 + threadIdx.x;   // 4*C*256*256
  long tot = (long)4 * C * 65536;
  if (idx >= tot) return;
  int x = idx & 255;
  int y = (int)(idx >> 8) & 255;
  int c = (int)((idx >> 16) % C);
  int b = (int)((idx >> 16) / C);
  dst[idx] = src[(((long)(b * 256) + y) * 256 + x) * C + c];
}

// Bilinear warp of last input frame by flow, then 1x1 fuse conv + sigmoid.
__global__ void warp_fuse(const float* __restrict__ xin, const float* __restrict__ frame,
                          const float* __restrict__ flow, const float* __restrict__ fw,
                          const float* __restrict__ fb, float* __restrict__ pred)
{
  int idx = blockIdx.x * 256 + threadIdx.x;          // 4*256*256
  int x = idx & 255;
  int y = (idx >> 8) & 255;
  int b = idx >> 16;
  const float* fl = flow + (long)idx * 2;
  float mx = (float)x + fl[0];
  float my = (float)y + fl[1];
  float fx = floorf(mx), fy = floorf(my);
  float wx = mx - fx, wy = my - fy;
  int x0 = (int)fx; x0 = x0 < 0 ? 0 : (x0 > 255 ? 255 : x0);
  int y0 = (int)fy; y0 = y0 < 0 ? 0 : (y0 > 255 ? 255 : y0);
  int x1 = x0 + 1 > 255 ? 255 : x0 + 1;
  int y1 = y0 + 1 > 255 ? 255 : y0 + 1;
  float in6[6];
  const float* fr = frame + (long)idx * 3;
  in6[0] = fr[0]; in6[1] = fr[1]; in6[2] = fr[2];
  for (int c = 0; c < 3; ++c) {
    const float* img = xin + (((long)(b * 3 + c) * 4 + 3) * 256) * 256;  // t = T-1
    float v00 = img[y0 * 256 + x0], v01 = img[y0 * 256 + x1];
    float v10 = img[y1 * 256 + x0], v11 = img[y1 * 256 + x1];
    in6[3 + c] = (1.f - wy) * ((1.f - wx) * v00 + wx * v01)
               +        wy  * ((1.f - wx) * v10 + wx * v11);
  }
  for (int co = 0; co < 3; ++co) {
    float s = fb[co];
    for (int ci = 0; ci < 6; ++ci) s += fw[co * 6 + ci] * in6[ci];
    pred[((long)(b * 3 + co) * 256 + y) * 256 + x] = 1.f / (1.f + expf(-s));
  }
}

// ---------------------------------------------------------------------------
extern "C" void kernel_launch(void* const* d_in, const int* in_sizes, int n_in,
                              void* d_out, int out_size, void* d_ws, size_t ws_size,
                              hipStream_t stream)
{
  (void)in_sizes; (void)n_in; (void)out_size; (void)ws_size;
  const float* x = (const float*)d_in[0];
  const float* cw[4]  = {(const float*)d_in[1], (const float*)d_in[2],
                         (const float*)d_in[3], (const float*)d_in[4]};
  const float* lw[3]  = {(const float*)d_in[5], (const float*)d_in[6], (const float*)d_in[7]};
  const float* lb[3]  = {(const float*)d_in[8], (const float*)d_in[9], (const float*)d_in[10]};
  const float* gfw[5] = {(const float*)d_in[11], (const float*)d_in[12], (const float*)d_in[13],
                         (const float*)d_in[14], (const float*)d_in[15]};
  const float* gfb[5] = {(const float*)d_in[16], (const float*)d_in[17], (const float*)d_in[18],
                         (const float*)d_in[19], (const float*)d_in[20]};
  const float* glw[5] = {(const float*)d_in[21], (const float*)d_in[22], (const float*)d_in[23],
                         (const float*)d_in[24], (const float*)d_in[25]};
  const float* glb[5] = {(const float*)d_in[26], (const float*)d_in[27], (const float*)d_in[28],
                         (const float*)d_in[29], (const float*)d_in[30]};
  const float* fw = (const float*)d_in[31];
  const float* fb = (const float*)d_in[32];

  char* p = (char*)d_ws;
  auto alloc = [&](size_t bytes) -> char* {
    char* r = p; p += (bytes + 255) & ~(size_t)255; return r;
  };
  f16*   e1p   = (f16*)alloc(17305600ULL * 2);   // (16,130,130,64)
  f16*   e2p   = (f16*)alloc(8921088ULL  * 2);   // (16,66,66,128)
  f16*   e3p   = (f16*)alloc(4734976ULL  * 2);   // (16,34,34,256)
  f16*   e4    = (f16*)alloc(2097152ULL  * 2);   // (16,16,16,512)
  f16*   xh    = (f16*)alloc(1327104ULL  * 2);   // (4,18,18,1024)
  f16*   hbuf  = (f16*)alloc(524288ULL   * 2);   // (4,16,16,512)
  f16*   seqA  = (f16*)alloc(2097152ULL  * 2);   // (4,4,16,16,512)
  f16*   seqB  = (f16*)alloc(2097152ULL  * 2);
  f16*   bufA  = (f16*)alloc(34076672ULL * 2);   // dilated inputs (max (4,258,258,128))
  f16*   bufB  = (f16*)alloc(17038336ULL * 2);   // gen activations (max (4,258,258,64))
  f16*   pkE2  = (f16*)alloc(131072ULL   * 2);
  f16*   pkE3  = (f16*)alloc(524288ULL   * 2);
  f16*   pkE4  = (f16*)alloc(2097152ULL  * 2);
  f16*   pkL   = (f16*)alloc(18874368ULL * 2);   // repacked per LSTM layer
  f16*   pkG   = (f16*)alloc(2359296ULL  * 2);   // repacked per gen layer
  float* gates = (float*)alloc(2097152ULL * 4);  // (4,16,16,2048)
  float* cbuf  = (float*)alloc(524288ULL  * 4);
  float* frameA= (float*)alloc(786432ULL  * 4);  // (4,256,256,3)
  float* flowA = (float*)alloc(524288ULL  * 4);  // (4,256,256,2)

  // zero padded-buffer borders (interiors are fully overwritten every call)
  (void)hipMemsetAsync(e1p, 0, 17305600ULL * 2, stream);
  (void)hipMemsetAsync(e2p, 0, 8921088ULL  * 2, stream);
  (void)hipMemsetAsync(e3p, 0, 4734976ULL  * 2, stream);
  (void)hipMemsetAsync(xh,  0, 1327104ULL  * 2, stream);

  // ---- encoder ----
  conv1_enc<<<65536, 256, 0, stream>>>(x, cw[0], e1p);
  pack_wmma_b<<<131072 / 256, 256, 0, stream>>>(cw[1], pkE2, 128,  64, 4, 4, 128);
  pack_wmma_b<<<524288 / 256, 256, 0, stream>>>(cw[2], pkE3, 256, 128, 4, 4, 256);
  pack_wmma_b<<<2097152 / 256, 256, 0, stream>>>(cw[3], pkE4, 512, 256, 4, 4, 512);
  conv_wmma<ACT_LEAKY, f16, 2, 4><<<dim3(256, 2), 256, 0, stream>>>(
      e1p, pkE2, nullptr, e2p, 130, 130, 64, 64, 64, 128, 128, 4, 4, 2, 66, 66, 1);
  conv_wmma<ACT_LEAKY, f16, 2, 4><<<dim3(64, 4), 256, 0, stream>>>(
      e2p, pkE3, nullptr, e3p, 66, 66, 128, 32, 32, 256, 256, 4, 4, 2, 34, 34, 1);
  conv_wmma<ACT_LEAKY, f16, 2, 4><<<dim3(16, 8), 256, 0, stream>>>(
      e3p, pkE4, nullptr, e4, 34, 34, 256, 16, 16, 512, 512, 4, 4, 2, 16, 16, 0);

  // ---- ConvLSTM (3 layers x 4 timesteps) ----
  const long FS = 16 * 16 * 512;
  for (int l = 0; l < 3; ++l) {
    pack_wmma_b<<<18874368 / 256, 256, 0, stream>>>(lw[l], pkL, 2048, 1024, 3, 3, 2048);
    (void)hipMemsetAsync(hbuf, 0, 524288ULL * 2, stream);
    (void)hipMemsetAsync(cbuf, 0, 524288ULL * 4, stream);
    f16* sOut = (l == 1) ? seqB : seqA;
    f16* sIn  = (l == 1) ? seqA : seqB;
    for (int t = 0; t < 4; ++t) {
      const f16* xt; long bStride;
      if (l == 0) { xt = e4 + (long)t * FS;      bStride = 4 * FS; }  // frame n=b*T+t
      else        { xt = sIn + (long)t * 4 * FS; bStride = FS; }      // seq [t][b]
      build_xh<<<1048576 / 256, 256, 0, stream>>>(xt, bStride, hbuf, xh);
      conv_wmma<ACT_NONE, float, 2, 4><<<dim3(4, 32), 256, 0, stream>>>(
          xh, pkL, lb[l], gates, 18, 18, 1024, 16, 16, 2048, 2048, 3, 3, 1, 16, 16, 0);
      lstm_point<<<524288 / 256, 256, 0, stream>>>(gates, cbuf, hbuf, sOut + (long)t * 4 * FS);
    }
  }
  const f16* out_me = seqA + 3 * 4 * FS;   // layer-2 output at t = T-1

  // ---- generators (deconv = zero-scatter + stride-1 WMMA conv) ----
  auto run_gen = [&](const float* const* ws, const float* const* bs,
                     int outCh, float* actOut, bool useSig) {
    const int  Hs[4] = {16, 32, 64, 128};
    const int ICs[4] = {512, 512, 256, 128};
    const int OCs[4] = {512, 256, 128, 64};
    const f16* prev = out_me;
    for (int i = 0; i < 4; ++i) {
      int H = Hs[i], IC = ICs[i], OC = OCs[i], S = 2 * H + 2, OH = 2 * H;
      long ptot = (long)(9 * IC / 32) * (OC / 16) * 512;
      pack_wmma_b<<<(unsigned)((ptot + 255) / 256), 256, 0, stream>>>(ws[i], pkG, OC, IC, 3, 3, OC);
      (void)hipMemsetAsync(bufA, 0, (size_t)4 * S * S * IC * 2, stream);
      long stot = (long)4 * H * H * IC;
      scatter_dil<<<(unsigned)((stot + 255) / 256), 256, 0, stream>>>(prev, bufA, H, IC);
      if (i == 3) (void)hipMemsetAsync(bufB, 0, 17038336ULL * 2, stream);  // g4 padded borders
      int M = 4 * OH * OH;
      int OHp = (i == 3) ? 258 : OH, padO = (i == 3) ? 1 : 0;
      conv_wmma<ACT_RELU, f16, 2, 4><<<dim3(M / 256, OC / 64), 256, 0, stream>>>(
          bufA, pkG, bs[i], bufB, S, S, IC, OH, OH, OC, OC, 3, 3, 1, OHp, OHp, padO);
      prev = bufB;
    }
    pack_wmma_b<<<(9216 + 255) / 256, 256, 0, stream>>>(ws[4], pkG, outCh, 64, 3, 3, 16);
    if (useSig)
      conv_wmma<ACT_SIG, float, 2, 1><<<dim3(1024, 1), 256, 0, stream>>>(
          bufB, pkG, bs[4], actOut, 258, 258, 64, 256, 256, outCh, 16, 3, 3, 1, 256, 256, 0);
    else
      conv_wmma<ACT_TANH, float, 2, 1><<<dim3(1024, 1), 256, 0, stream>>>(
          bufB, pkG, bs[4], actOut, 258, 258, 64, 256, 256, outCh, 16, 3, 3, 1, 256, 256, 0);
  };
  run_gen(gfw, gfb, 3, frameA, true);
  run_gen(glw, glb, 2, flowA, false);

  // ---- outputs: frame_pred, flow_pred, prediction (NCHW f32, concatenated) ----
  float* outF = (float*)d_out;
  write_nchw<<<(4 * 3 * 65536) / 256, 256, 0, stream>>>(frameA, outF, 3);
  write_nchw<<<(4 * 2 * 65536) / 256, 256, 0, stream>>>(flowA, outF + 786432, 2);
  warp_fuse<<<(4 * 65536) / 256, 256, 0, stream>>>(x, frameA, flowA, fw, fb,
                                                   outF + 786432 + 524288);
}
